// SymmetricContraction_50843822850736
// MI455X (gfx1250) — compile-verified, matrix operand order checked by reading, not run on hardware
//
#include <hip/hip_runtime.h>

// ============================================================================
// MACE-style symmetric contraction on gfx1250 (MI455X), WMMA + async-LDS.
//
//   T3[(k,i),n] = sum_{p,q,j} u3[p,q,j,k,i] x_p x_q x_j   (per column n=(b,c))
//   out[n,i]    = sum_k T3 w3[s,k,c] + sum_k T2 w2[s,k,c] + T1 w1[s,0,c]
//
// Heavy stage as GEMM:  Y[(m,p), n] = U[(m,p),(q,j)] @ XX[(q,j), n]
//   U rows: 80*16 order-3 rows + 16 order-2 rows = 1296 rows = 81 M-tiles
//   K = 256 -> 8 x v_wmma_f32_16x16x32_f16 per M-tile
// then stage-B (VALU): T3[m,n] = sum_p Y[(m,p),n] * x_p[n].
//
// U (648KB, fp16, WMMA A-register layout) is staged through LDS per workgroup
// with double-buffered async global->LDS DMA (ASYNCcnt). Inside a chunk the
// A-operands are software-pipelined from LDS with prefetch distance 2, and
// stage-B consumption is deferred one group (ping-pong accumulators) so the
// WMMA->VALU hazard window is filled by the next group's WMMAs.
// Workspace needed: 81*8*32*16*2 = 663,552 bytes.
// ============================================================================

typedef __attribute__((ext_vector_type(16))) _Float16 v16h;
typedef __attribute__((ext_vector_type(8)))  float    v8f;
typedef int i32x4 __attribute__((vector_size(16)));

#define NB     512
#define NC     128
#define ND     16
#define NCOLS  (NB*NC)   // 65536
#define MUL3   20
#define NTILES 81        // 80 order-3 tiles (t*4+g) + 1 order-2 tile
#define KST    8         // 256 / 32
#define TILE_BYTES  8192          // 8 ksteps * 32 lanes * 32 B
#define TILE_HALFS  4096
#define CHUNK_TILES 4             // one k-index x 4 output components
#define CHUNK_HALFS (CHUNK_TILES*TILE_HALFS)   // 16384 (32 KB)
#define NFULL_CH    20            // chunks 0..19 = tiles 0..79; chunk 20 = tile 80

#ifndef __has_builtin
#define __has_builtin(x) 0
#endif
#if __has_builtin(__builtin_amdgcn_global_load_async_to_lds_b128)
#define HAVE_ASYNC_LDS 1
#else
#define HAVE_ASYNC_LDS 0
#endif

// ISA 16-bit A-matrix 16x32 layout: element e of v16h <-> K index.
__host__ __device__ __forceinline__ constexpr int kmap(int e, int hi) {
  const int v = e >> 1, w = e & 1;
  const int base = (v < 4) ? (2 * v) : (16 + 2 * (v - 4));
  return base + w + hi * 8;
}

// ----------------------------------------------------------------------------
// Pack U into WMMA-A register layout, fp16. Tile order: tl<80 -> t=tl>>2,
// g=tl&3 (g==0: 0e k=t; g>0: 1o comp g-1, k=t). tl==80 -> order-2 rows.
// ----------------------------------------------------------------------------
__global__ void pack_u_kernel(const float* __restrict__ u2_0e,
                              const float* __restrict__ u3_0e,
                              const float* __restrict__ u2_1o,
                              const float* __restrict__ u3_1o,
                              _Float16* __restrict__ Uh) {
  const int gid  = blockIdx.x * 256 + threadIdx.x;  // exactly NTILES*KST*32
  const int lane = gid & 31;
  const int st   = (gid >> 5) & 7;
  const int tile = gid >> 8;
  const int M    = lane & 15;          // row within tile
  const int hi   = (lane >> 4) & 1;    // K-half select
  v16h hv;
#pragma unroll
  for (int e = 0; e < 16; ++e) {
    const int kg = st * 32 + kmap(e, hi);   // 0..255 over (q,j)
    const int a  = kg >> 4;                 // q
    const int b  = kg & 15;                 // j
    float u;
    if (tile < 80) {
      const int t = tile >> 2, g = tile & 3;
      if (g == 0) u = u3_0e[((M * 16 + a) * 16 + b) * MUL3 + t];
      else        u = u3_1o[(((M * 16 + a) * 16 + b) * MUL3 + t) * 3 + (g - 1)];
    } else {                                // order-2 rows
      if (M < 4) {
        u = u2_0e[(a * 16 + b) * 4 + M];
      } else {
        const int k2 = (M - 4) / 3, i = (M - 4) % 3;
        u = u2_1o[((a * 16 + b) * 4 + k2) * 3 + i];
      }
    }
    hv[e] = (_Float16)u;
  }
  *(v16h*)(Uh + (size_t)gid * 16) = hv;
}

// ----------------------------------------------------------------------------
// Async (or fallback) copy of one chunk of U into an LDS buffer.
// Full chunks: 32 KB (8 sweeps of 256 lanes x 16B); last chunk: 8 KB.
// ----------------------------------------------------------------------------
__device__ __forceinline__ void issue_copy(const _Float16* __restrict__ Uh,
                                           int ch, _Float16* lbuf, int tid) {
  const char* gsrc = (const char*)Uh + (size_t)ch * CHUNK_TILES * TILE_BYTES;
  char*       ldst = (char*)lbuf;
  const int   iters = (ch == NFULL_CH) ? 2 : 8;   // 8KB vs 32KB
  for (int i = 0; i < iters; ++i) {
    const int off = (i * 256 + tid) * 16;
#if HAVE_ASYNC_LDS
    __builtin_amdgcn_global_load_async_to_lds_b128(
        (__attribute__((address_space(1))) i32x4*)(gsrc + off),
        (__attribute__((address_space(3))) i32x4*)(ldst + off),
        0, 0);
#else
    *(float4*)(ldst + off) = *(const float4*)(gsrc + off);
#endif
  }
}

__device__ __forceinline__ void wait_async_copies() {
#if HAVE_ASYNC_LDS
#if __has_builtin(__builtin_amdgcn_s_wait_asynccnt)
  __builtin_amdgcn_s_wait_asynccnt(0);
#else
  asm volatile("s_wait_asynccnt 0" ::: "memory");
#endif
#endif
}

// ----------------------------------------------------------------------------
// Main kernel: 8 waves/WG, each wave owns 32 columns (two 16-col B-tiles).
// U chunks double-buffered in LDS; WMMAs consume chunk ch while the async DMA
// fills chunk ch+1. No divergence before WMMA (EXEC all ones).
// ----------------------------------------------------------------------------
#define WMMA_F16(A, Bm, Cm) \
  __builtin_amdgcn_wmma_f32_16x16x32_f16(false, (A), false, (Bm), (short)0, (Cm), false, false)

__global__ __launch_bounds__(256) void symcon_main(
    const float* __restrict__ x, const int* __restrict__ sp_index,
    const float* __restrict__ u1_0e, const float* __restrict__ w1_0e,
    const float* __restrict__ w2_0e, const float* __restrict__ w3_0e,
    const float* __restrict__ u1_1o, const float* __restrict__ w1_1o,
    const float* __restrict__ w2_1o, const float* __restrict__ w3_1o,
    const _Float16* __restrict__ Uh, float* __restrict__ out) {
  __shared__ __align__(16) _Float16 smem[2][CHUNK_HALFS];   // 2 x 32 KB

  const int tid  = threadIdx.x;
  const int l    = tid & 31;
  const int col  = l & 15;
  const int hi   = l >> 4;
  const int wave = blockIdx.x * (blockDim.x >> 5) + (tid >> 5);
  const int n0   = wave * 32;
  const int nA   = n0 + col;        // column for B-tile A
  const int nB   = n0 + 16 + col;   // column for B-tile B

  // kick off chunk 0 DMA before doing per-column setup (overlap)
  issue_copy(Uh, 0, &smem[0][0], tid);

  // ---- load x for both columns (16 floats each) -----------------------------
  float xfA[16], xfB[16];
  {
    const float4* pa = (const float4*)(x + (size_t)nA * ND);
    const float4* pb = (const float4*)(x + (size_t)nB * ND);
#pragma unroll
    for (int q = 0; q < 4; ++q) {
      const float4 ta = pa[q], tb = pb[q];
      xfA[4 * q + 0] = ta.x; xfA[4 * q + 1] = ta.y;
      xfA[4 * q + 2] = ta.z; xfA[4 * q + 3] = ta.w;
      xfB[4 * q + 0] = tb.x; xfB[4 * q + 1] = tb.y;
      xfB[4 * q + 2] = tb.z; xfB[4 * q + 3] = tb.w;
    }
  }
  const int cA = nA & (NC - 1), cB = nB & (NC - 1);
  const int sA = sp_index[nA >> 7], sB = sp_index[nB >> 7];

  // ---- order-1 moments ------------------------------------------------------
  float t1A0 = 0.f, t1B0 = 0.f, t1A1[3] = {0, 0, 0}, t1B1[3] = {0, 0, 0};
#pragma unroll
  for (int p = 0; p < 16; ++p) {
    const float ua = u1_0e[p];
    t1A0 += ua * xfA[p]; t1B0 += ua * xfB[p];
#pragma unroll
    for (int i = 0; i < 3; ++i) {
      const float ub = u1_1o[p * 3 + i];
      t1A1[i] += ub * xfA[p]; t1B1[i] += ub * xfB[p];
    }
  }
  float oA[4], oB[4];
  {
    const float wa0 = w1_0e[sA * NC + cA], wb0 = w1_0e[sB * NC + cB];
    const float wa1 = w1_1o[sA * NC + cA], wb1 = w1_1o[sB * NC + cB];
    oA[0] = t1A0 * wa0; oB[0] = t1B0 * wb0;
#pragma unroll
    for (int i = 0; i < 3; ++i) { oA[1 + i] = t1A1[i] * wa1; oB[1 + i] = t1B1[i] * wb1; }
  }

  // ---- B operands: XX[(q,j)] = x_q * x_j in WMMA-B register layout ----------
  v16h bA[KST], bB[KST];
#pragma unroll
  for (int st = 0; st < KST; ++st) {
#pragma unroll
    for (int e = 0; e < 16; ++e) {
      const int klo = st * 32 + kmap(e, 0), khi = st * 32 + kmap(e, 1);
      const int alo = klo >> 4, blo = klo & 15;
      const int ahi = khi >> 4, bhi = khi & 15;
      const float vA = hi ? (xfA[ahi] * xfA[bhi]) : (xfA[alo] * xfA[blo]);
      const float vB = hi ? (xfB[ahi] * xfB[bhi]) : (xfB[alo] * xfB[blo]);
      bA[st][e] = (_Float16)vA;
      bB[st][e] = (_Float16)vB;
    }
  }

  // stage-B x values: C/D layout -> VGPR v holds row p = v + 8*hi
  float xsA[8], xsB[8];
#pragma unroll
  for (int v = 0; v < 8; ++v) {
    xsA[v] = hi ? xfA[v + 8] : xfA[v];
    xsB[v] = hi ? xfB[v + 8] : xfB[v];
  }

  // ---- order-3 chunks: chunk ch = k-index t=ch, groups g=0..3 ---------------
  for (int ch = 0; ch < NFULL_CH; ++ch) {
    wait_async_copies();
    __syncthreads();                               // chunk ch resident in LDS
    issue_copy(Uh, ch + 1, &smem[(ch + 1) & 1][0], tid);   // prefetch next

    const v16h* bufv = (const v16h*)&smem[ch & 1][0];
    const float cfA0 = w3_0e[(sA * MUL3 + ch) * NC + cA];
    const float cfB0 = w3_0e[(sB * MUL3 + ch) * NC + cB];
    const float cfA1 = w3_1o[(sA * MUL3 + ch) * NC + cA];
    const float cfB1 = w3_1o[(sB * MUL3 + ch) * NC + cB];

    // stage-B: T3 = sum_p Y[p,n]*x_p, fold lane halves, accumulate output
    auto stageB = [&](int g, const v8f& aA, const v8f& aB) {
      float pA = 0.f, pB = 0.f;
#pragma unroll
      for (int v = 0; v < 8; ++v) { pA += aA[v] * xsA[v]; pB += aB[v] * xsB[v]; }
      pA += __shfl_xor(pA, 16, 32);
      pB += __shfl_xor(pB, 16, 32);
      const int io = (g == 0) ? 0 : g;
      oA[io] += pA * ((g == 0) ? cfA0 : cfA1);
      oB[io] += pB * ((g == 0) ? cfB0 : cfB1);
    };

    // flattened 32-step WMMA loop: A prefetch distance 2, ping-pong acc,
    // stage-B of group g-1 deferred until 4 WMMAs of group g are in flight.
    v16h a0 = bufv[0 * 32 + l];
    v16h a1 = bufv[1 * 32 + l];
    v8f accA[2], accB[2];
#pragma unroll
    for (int u = 0; u < 32; ++u) {
      const int g = u >> 3, st = u & 7, gp = g & 1;
      if (st == 0) {
        accA[gp] = (v8f){0.f, 0.f, 0.f, 0.f, 0.f, 0.f, 0.f, 0.f};
        accB[gp] = (v8f){0.f, 0.f, 0.f, 0.f, 0.f, 0.f, 0.f, 0.f};
      }
      const v16h ac = (u & 1) ? a1 : a0;
      if (u + 2 < 32) {                    // prefetch k-step u+2
        const v16h an = bufv[(u + 2) * 32 + l];
        if (u & 1) a1 = an; else a0 = an;
      }
      accA[gp] = WMMA_F16(ac, bA[st], accA[gp]);
      accB[gp] = WMMA_F16(ac, bB[st], accB[gp]);
      if (st == 1 && g >= 1) stageB(g - 1, accA[gp ^ 1], accB[gp ^ 1]);
    }
    stageB(3, accA[1], accB[1]);
    __syncthreads();                               // release buffer ch&1
  }

  // ---- order-2 tile (chunk 20, single tile, in buf 0) -----------------------
  {
    wait_async_copies();
    __syncthreads();
    const v16h* bufv = (const v16h*)&smem[0][0];
    v16h a0 = bufv[0 * 32 + l];
    v16h a1 = bufv[1 * 32 + l];
    v8f accA = {0.f, 0.f, 0.f, 0.f, 0.f, 0.f, 0.f, 0.f};
    v8f accB = {0.f, 0.f, 0.f, 0.f, 0.f, 0.f, 0.f, 0.f};
#pragma unroll
    for (int u = 0; u < KST; ++u) {
      const v16h ac = (u & 1) ? a1 : a0;
      if (u + 2 < KST) {
        const v16h an = bufv[(u + 2) * 32 + l];
        if (u & 1) a1 = an; else a0 = an;
      }
      accA = WMMA_F16(ac, bA[u], accA);
      accB = WMMA_F16(ac, bB[u], accB);
    }
    float opA[4] = {0, 0, 0, 0}, opB[4] = {0, 0, 0, 0};
#pragma unroll
    for (int v = 0; v < 8; ++v) {
      const int rlo = v, rhi = v + 8;              // rows per lane half (CT)
      const int klo = (rlo < 4) ? rlo : (rlo - 4) / 3;
      const int ilo = (rlo < 4) ? 0 : 1 + (rlo - 4) % 3;
      const int khi = (rhi - 4) / 3;
      const int ihi = 1 + (rhi - 4) % 3;
      const float* wlo = (rlo < 4) ? w2_0e : w2_1o;
      const float cfA = hi ? w2_1o[(sA * 4 + khi) * NC + cA]
                           : wlo[(sA * 4 + klo) * NC + cA];
      const float cfB = hi ? w2_1o[(sB * 4 + khi) * NC + cB]
                           : wlo[(sB * 4 + klo) * NC + cB];
      const float tA = accA[v] * cfA, tB = accB[v] * cfB;
      if (ilo == ihi) { opA[ilo] += tA; opB[ilo] += tB; }
      else {
        opA[ilo] += hi ? 0.f : tA;  opA[ihi] += hi ? tA : 0.f;
        opB[ilo] += hi ? 0.f : tB;  opB[ihi] += hi ? tB : 0.f;
      }
    }
#pragma unroll
    for (int i = 0; i < 4; ++i) {                  // fold lane halves
      oA[i] += opA[i] + __shfl_xor(opA[i], 16, 32);
      oB[i] += opB[i] + __shfl_xor(opB[i], 16, 32);
    }
  }

  // ---- store (B,C,4) --------------------------------------------------------
  if (!hi) {
    const float4 ra = {oA[0], oA[1], oA[2], oA[3]};
    const float4 rb = {oB[0], oB[1], oB[2], oB[3]};
    *(float4*)(out + (size_t)nA * 4) = ra;
    *(float4*)(out + (size_t)nB * 4) = rb;
  }
}

// ----------------------------------------------------------------------------
extern "C" void kernel_launch(void* const* d_in, const int* in_sizes, int n_in,
                              void* d_out, int out_size, void* d_ws, size_t ws_size,
                              hipStream_t stream) {
  (void)in_sizes; (void)n_in; (void)out_size; (void)ws_size;
  // setup_inputs() dict order:
  const float* x      = (const float*)d_in[0];
  const int*   idx    = (const int*)  d_in[1];
  const float* u1_0e  = (const float*)d_in[2];
  const float* w1_0e  = (const float*)d_in[3];
  const float* u2_0e  = (const float*)d_in[4];
  const float* w2_0e  = (const float*)d_in[5];
  const float* u3_0e  = (const float*)d_in[6];
  const float* w3_0e  = (const float*)d_in[7];
  const float* u1_1o  = (const float*)d_in[8];
  const float* w1_1o  = (const float*)d_in[9];
  const float* u2_1o  = (const float*)d_in[10];
  const float* w2_1o  = (const float*)d_in[11];
  const float* u3_1o  = (const float*)d_in[12];
  const float* w3_1o  = (const float*)d_in[13];

  _Float16* Uh = (_Float16*)d_ws;   // 663,552 bytes

  // 1) repack basis into WMMA-A fp16 layout (81 tiles * 8 ksteps * 32 lanes)
  pack_u_kernel<<<NTILES, 256, 0, stream>>>(u2_0e, u3_0e, u2_1o, u3_1o, Uh);

  // 2) main contraction: 256 WGs x 8 waves x 32 columns = 65536 columns
  symcon_main<<<NCOLS / 256, 256, 0, stream>>>(
      x, idx, u1_0e, w1_0e, w2_0e, w3_0e, u1_1o, w1_1o, w2_1o, w3_1o,
      (const _Float16*)Uh, (float*)d_out);
}
